// BiLSTM_CRF_L_87651692576922
// MI455X (gfx1250) — matrix operand order repaired and checked by reading, not compiled, not run on hardware
//
#include <hip/hip_runtime.h>

#define Lx 256
#define Bx 128
#define Ex 256
#define HDx 256
#define G4 1024   // 4*HD
#define Hx 512
#define Kt 32
#define KK 1024   // K*K

typedef __attribute__((ext_vector_type(16))) __bf16 v16bf;
typedef __attribute__((ext_vector_type(8)))  float  v8f;

union fragbf { v16bf v; uint4 q[2]; };

__device__ __forceinline__ unsigned short f2bf(float f) {
    unsigned u = __float_as_uint(f);
    unsigned r = u + 0x7FFFu + ((u >> 16) & 1u);   // RNE
    return (unsigned short)(r >> 16);
}
__device__ __forceinline__ float sigm(float x) { return 1.f / (1.f + __expf(-x)); }

// ---------------- utility kernels ----------------

__global__ void zero_kernel(unsigned int* p, int n) {
    int i = blockIdx.x * blockDim.x + threadIdx.x;
    if (i < n) p[i] = 0u;
}

__global__ void cast_f32_bf16(const float* src, unsigned short* dst, int n) {
    int i = blockIdx.x * blockDim.x + threadIdx.x;
    if (i < n) dst[i] = f2bf(src[i]);
}

__global__ void embed_bf16(const int* sent, const float* emb, unsigned short* xbf) {
    int row = blockIdx.x;            // l*B + b
    int e   = threadIdx.x;           // 0..255
    int v   = sent[row];
    xbf[(size_t)row * Ex + e] = f2bf(emb[(size_t)v * Ex + e]);
}

__global__ void concat_bf16(const float* hf, const float* hb, unsigned short* hcat) {
    int row = blockIdx.x;
    int e   = threadIdx.x;           // 256
    hcat[(size_t)row * Hx + e]       = f2bf(hf[(size_t)row * HDx + e]);
    hcat[(size_t)row * Hx + 256 + e] = f2bf(hb[(size_t)row * HDx + e]);
}

// ---- bf16 WMMA GEMM: C[M,N] = A[M,K] @ W[N,K]^T + bias0 + bias1 ----
// 256 threads (8 waves); block tile 16(M) x 128(N); all fragment feeds are b128.

template <int KDIM>
__global__ __launch_bounds__(256) void gemm_bf16(const unsigned short* __restrict__ A,
                                                 const unsigned short* __restrict__ W,
                                                 const float* bias0, const float* bias1,
                                                 float* __restrict__ C, int M, int N) {
    const int K  = KDIM;
    const int KP = KDIM + 8;                       // keeps rows 16B-aligned + bank spread
    __shared__ unsigned short Asm[16 * (KDIM + 8)];
    int m0    = blockIdx.y * 16;
    int n0blk = blockIdx.x * 128;
    int tid   = threadIdx.x;

    const int rowU = K / 8;                        // uint4 units per row
    for (int u = tid; u < 16 * rowU; u += 256) {
        int r = u / rowU, c8 = u % rowU;
        *(uint4*)(Asm + r * KP + c8 * 8) =
            *(const uint4*)(A + (size_t)(m0 + r) * K + c8 * 8);
    }
    __syncthreads();

    int wave = tid >> 5, lane = tid & 31;
    int hi = lane >> 4, ln = lane & 15;
    int n0 = n0blk + wave * 16;

    v8f acc = {0.f, 0.f, 0.f, 0.f, 0.f, 0.f, 0.f, 0.f};
    for (int kk = 0; kk < K; kk += 32) {
        fragbf a, b;
        int arow = ln * KP + kk + hi * 8;          // A 16x32 bf16: k = e(+8) + hi*8
        a.q[0] = *(const uint4*)(Asm + arow);
        a.q[1] = *(const uint4*)(Asm + arow + 16);
        const unsigned short* wp = W + (size_t)(n0 + ln) * K + kk + hi * 8;  // B 32x16
        b.q[0] = *(const uint4*)(wp);
        b.q[1] = *(const uint4*)(wp + 16);
        acc = __builtin_amdgcn_wmma_f32_16x16x32_bf16(false, a.v, false, b.v,
                                                      (short)0, acc, false, false);
    }
    float bn = (bias0 ? bias0[n0 + ln] : 0.f) + (bias1 ? bias1[n0 + ln] : 0.f);
#pragma unroll
    for (int r = 0; r < 8; ++r) {
        int m = m0 + r + hi * 8;                   // C/D fragment layout
        C[(size_t)m * N + (n0 + ln)] = acc[r] + bn;
    }
}

// ---------------- fused persistent BiLSTM recurrence ----------------
// grid (8, 2): y = direction, x = 32-wide hidden slice. 256 thr = 8 waves.
// Recurrent GEMM in bf16 WMMA (16x16x32), c-state fp32 in registers,
// h history kept in fp32 (for projection) + bf16 (for next-step GEMM).

__global__ __launch_bounds__(256) void lstm_kernel(const float* __restrict__ gates_f,
                                                   const float* __restrict__ gates_b,
                                                   const float* __restrict__ Whh_f,
                                                   const float* __restrict__ Whh_b,
                                                   float* hf, float* hb,
                                                   unsigned short* hbf_f,
                                                   unsigned short* hbf_b,
                                                   unsigned int* counters) {
    const int SP = HDx + 8;                        // 264 ushorts/row
    extern __shared__ unsigned short smem_us[];
    unsigned short* hsm = smem_us;                 // [128][264] bf16 h_prev
    unsigned short* wsm = smem_us + Bx * SP;       // [128][264] bf16 Whh slice

    int d  = blockIdx.y;
    int wg = blockIdx.x;                           // 0..7
    const float* gates = d ? gates_b : gates_f;
    const float* Whh   = d ? Whh_b  : Whh_f;
    float*          hout = d ? hb    : hf;
    unsigned short* hbf  = d ? hbf_b : hbf_f;
    unsigned int*   cnt  = counters + d * Lx;
    int j0  = wg * 32;
    int tid = threadIdx.x;
    int wave = tid >> 5, lane = tid & 31, hi = lane >> 4, ln = lane & 15;
    int mi = wave;

    // stage + convert this WG's 128 Whh rows once:
    // local col nl -> global row (nl>>5)*256 + j0 + (nl&31)
    for (int idx = tid; idx < 128 * HDx; idx += 256) {
        int nl = idx >> 8;
        int k  = idx & 255;
        int n  = (nl >> 5) * 256 + j0 + (nl & 31);
        wsm[nl * SP + k] = f2bf(Whh[(size_t)n * HDx + k]);
    }

    float creg[16];
#pragma unroll
    for (int i = 0; i < 16; ++i) creg[i] = 0.f;

    for (int s = 0; s < Lx; ++s) {
        int t     = d ? (Lx - 1 - s) : s;
        int tprev = d ? (t + 1) : (t - 1);

        // prefetch this step's gate rows (wave covers 16 m-rows x 4 quadrants)
        {
            int pm = mi * 16 + (lane >> 2);
            int pq = lane & 3;
            size_t pb = ((size_t)t * Bx + pm) * G4 + pq * 256 + j0;
            __builtin_prefetch(&gates[pb], 0, 1);
            __builtin_prefetch(&gates[pb + (size_t)8 * G4], 0, 1);
        }

        v8f acc[8];
#pragma unroll
        for (int q = 0; q < 8; ++q) { v8f z = {0,0,0,0,0,0,0,0}; acc[q] = z; }

        if (s > 0) {  // stage bf16 h_prev: 4096 x uint4 (b128)
            const uint4* src = (const uint4*)(hbf + (size_t)tprev * Bx * HDx);
            for (int u = tid; u < (Bx * HDx) / 8; u += 256) {
                int row = u >> 5;                  // 32 uint4 per 256-elem row
                int c8  = u & 31;
                *(uint4*)(hsm + row * SP + c8 * 8) = src[u];
            }
        }
        __syncthreads();

        if (s > 0) {
            int arow = (mi * 16 + ln) * SP;
            for (int kk = 0; kk < HDx; kk += 32) {
                fragbf a;
                a.q[0] = *(const uint4*)(hsm + arow + kk + hi * 8);
                a.q[1] = *(const uint4*)(hsm + arow + kk + hi * 8 + 16);
#pragma unroll
                for (int nt = 0; nt < 8; ++nt) {
                    fragbf b;
                    int brow = (nt * 16 + ln) * SP + kk + hi * 8;
                    b.q[0] = *(const uint4*)(wsm + brow);
                    b.q[1] = *(const uint4*)(wsm + brow + 16);
                    acc[nt] = __builtin_amdgcn_wmma_f32_16x16x32_bf16(
                        false, a.v, false, b.v, (short)0, acc[nt], false, false);
                }
            }
        }

        // gates = acc + precomputed (x@Wih^T + bih + bhh); LSTM cell update
#pragma unroll
        for (int half = 0; half < 2; ++half) {
#pragma unroll
            for (int r = 0; r < 8; ++r) {
                int m = mi * 16 + r + hi * 8;
                int j = j0 + half * 16 + ln;
                size_t base = ((size_t)t * Bx + m) * G4;
                float iv = acc[0 * 2 + half][r] + gates[base + 0 * 256 + j];
                float fv = acc[1 * 2 + half][r] + gates[base + 1 * 256 + j];
                float gv = acc[2 * 2 + half][r] + gates[base + 2 * 256 + j];
                float ov = acc[3 * 2 + half][r] + gates[base + 3 * 256 + j];
                float c  = creg[half * 8 + r];
                float cn = sigm(fv) * c + sigm(iv) * tanhf(gv);
                float hn = sigm(ov) * tanhf(cn);
                creg[half * 8 + r] = cn;
                size_t ho = ((size_t)t * Bx + m) * HDx + j;
                hout[ho] = hn;
                hbf[ho]  = f2bf(hn);
            }
        }

        __threadfence();
        __syncthreads();
        if (s < Lx - 1) {   // cross-WG barrier (8 WGs per direction)
            if (tid == 0) {
                atomicAdd(&cnt[s], 1u);
                while (atomicAdd(&cnt[s], 0u) < 8u) { }
            }
            __syncthreads();
            __threadfence();
        }
    }
}

// ---------------- gold score: masked gather + deterministic reduction --------

__global__ __launch_bounds__(256) void gold_kernel(const float* __restrict__ scores,
                                                   const int* __restrict__ targets,
                                                   const unsigned char* __restrict__ masks,
                                                   float* part) {
    __shared__ float red[256];
    int i = blockIdx.x * 256 + threadIdx.x;    // 0..32767
    float v = 0.f;
    int tg = targets[i];
    if (masks[i]) v = scores[(size_t)i * KK + tg];
    red[threadIdx.x] = v;
    __syncthreads();
    for (int o = 128; o > 0; o >>= 1) {
        if (threadIdx.x < o) red[threadIdx.x] += red[threadIdx.x + o];
        __syncthreads();
    }
    if (threadIdx.x == 0) part[blockIdx.x] = red[0];
}

// ---------------- CRF forward: block per batch, wave per dest tag ----------

__global__ __launch_bounds__(1024) void crf_kernel(const float* __restrict__ scores,
                                                   const unsigned char* __restrict__ masks,
                                                   float* part) {
    __shared__ float dp[Kt];
    int b   = blockIdx.x;
    int tid = threadIdx.x;
    int j   = tid >> 5;    // dest tag (wave)
    int i   = tid & 31;    // source tag (lane)
    if (tid < Kt) dp[tid] = scores[(size_t)b * KK + tid];   // START_TAG row
    __syncthreads();
    for (int t = 1; t < Lx; ++t) {
        float s = scores[(((size_t)t * Bx) + b) * KK + i * Kt + j] + dp[i];
        float m = s;
#pragma unroll
        for (int o = 16; o > 0; o >>= 1) m = fmaxf(m, __shfl_xor(m, o, 32));
        float p = __expf(s - m);
#pragma unroll
        for (int o = 16; o > 0; o >>= 1) p += __shfl_xor(p, o, 32);
        float r = m + __logf(p);
        bool upd = masks[t * Bx + b] != 0;
        __syncthreads();
        if (i == 0 && upd) dp[j] = r;
        __syncthreads();
    }
    if (tid == 1) part[b] = dp[1];   // END_TAG = 1
}

__global__ void finalize_kernel(const float* gold_part, const float* crf_part, float* out) {
    __shared__ float g[128], f[128];
    int t = threadIdx.x;
    g[t] = gold_part[t]; f[t] = crf_part[t];
    __syncthreads();
    for (int o = 64; o > 0; o >>= 1) {
        if (t < o) { g[t] += g[t + o]; f[t] += f[t + o]; }
        __syncthreads();
    }
    if (t == 0) out[0] = (f[0] - g[0]) / (float)Bx;
}

// ---------------- host launch ----------------

extern "C" void kernel_launch(void* const* d_in, const int* in_sizes, int n_in,
                              void* d_out, int out_size, void* d_ws, size_t ws_size,
                              hipStream_t stream) {
    const int*   sent  = (const int*)d_in[0];
    const int*   targ  = (const int*)d_in[1];
    const unsigned char* masks = (const unsigned char*)d_in[2];
    const float* emb   = (const float*)d_in[3];
    const float* Wih_f = (const float*)d_in[4];
    const float* Whh_f = (const float*)d_in[5];
    const float* bih_f = (const float*)d_in[6];
    const float* bhh_f = (const float*)d_in[7];
    const float* Wih_b = (const float*)d_in[8];
    const float* Whh_b = (const float*)d_in[9];
    const float* bih_b = (const float*)d_in[10];
    const float* bhh_b = (const float*)d_in[11];
    const float* Wout  = (const float*)d_in[12];
    const float* bout  = (const float*)d_in[13];
    float* out = (float*)d_out;

    const size_t LB = (size_t)Lx * Bx;   // 32768
    char* ws = (char*)d_ws;
    size_t off = 0;
    auto take = [&](size_t bytes) { size_t r = off; off += (bytes + 255) & ~(size_t)255; return r; };
    size_t o_xbf   = take(LB * Ex * 2);
    size_t o_wihf  = take((size_t)G4 * Ex * 2);
    size_t o_wihb  = take((size_t)G4 * Ex * 2);
    size_t o_wout  = take((size_t)KK * Hx * 2);
    size_t o_gf    = take(LB * G4 * 4);
    size_t o_gb    = take(LB * G4 * 4);
    size_t o_hf    = take(LB * HDx * 4);
    size_t o_hb    = take(LB * HDx * 4);
    size_t o_hbf_f = take(LB * HDx * 2);
    size_t o_hbf_b = take(LB * HDx * 2);
    size_t o_hcat  = take(LB * Hx * 2);
    size_t o_sc    = take(LB * KK * 4);
    size_t o_cnt   = take(2 * Lx * 4);
    size_t o_gldp  = take(128 * 4);
    size_t o_crfp  = take(128 * 4);

    unsigned short* xbf   = (unsigned short*)(ws + o_xbf);
    unsigned short* wihfb = (unsigned short*)(ws + o_wihf);
    unsigned short* wihbb = (unsigned short*)(ws + o_wihb);
    unsigned short* woutb = (unsigned short*)(ws + o_wout);
    float* gates_f = (float*)(ws + o_gf);
    float* gates_b = (float*)(ws + o_gb);
    float* hf      = (float*)(ws + o_hf);
    float* hb      = (float*)(ws + o_hb);
    unsigned short* hbf_f = (unsigned short*)(ws + o_hbf_f);
    unsigned short* hbf_b = (unsigned short*)(ws + o_hbf_b);
    unsigned short* hcat  = (unsigned short*)(ws + o_hcat);
    float* scores  = (float*)(ws + o_sc);
    unsigned int* counters = (unsigned int*)(ws + o_cnt);
    float* gold_part = (float*)(ws + o_gldp);
    float* crf_part  = (float*)(ws + o_crfp);

    // zero counters + partial buffers every call
    int nz = (int)((o_crfp + 128 * 4 - o_cnt) / 4);
    zero_kernel<<<(nz + 255) / 256, 256, 0, stream>>>(counters, nz);

    // weight casts to bf16 (Whh is converted inside the LSTM kernel)
    cast_f32_bf16<<<(G4 * Ex + 255) / 256, 256, 0, stream>>>(Wih_f, wihfb, G4 * Ex);
    cast_f32_bf16<<<(G4 * Ex + 255) / 256, 256, 0, stream>>>(Wih_b, wihbb, G4 * Ex);
    cast_f32_bf16<<<(KK * Hx + 255) / 256, 256, 0, stream>>>(Wout, woutb, KK * Hx);

    // embedding gather -> bf16 x
    embed_bf16<<<(unsigned)LB, 256, 0, stream>>>(sent, emb, xbf);

    // time-parallel input GEMMs: gates = x @ Wih^T + bih + bhh
    gemm_bf16<Ex><<<dim3(KK / 128, (unsigned)(LB / 16)), 256, 0, stream>>>(
        xbf, wihfb, bih_f, bhh_f, gates_f, (int)LB, KK);
    gemm_bf16<Ex><<<dim3(KK / 128, (unsigned)(LB / 16)), 256, 0, stream>>>(
        xbf, wihbb, bih_b, bhh_b, gates_b, (int)LB, KK);

    // fused bidirectional recurrence (persistent, bf16 WMMA, fp32 cell state)
    size_t lstm_lds = (size_t)2 * Bx * (HDx + 8) * sizeof(unsigned short);  // 135168 B
    hipFuncSetAttribute((const void*)lstm_kernel,
                        hipFuncAttributeMaxDynamicSharedMemorySize, (int)lstm_lds);
    lstm_kernel<<<dim3(8, 2), 256, lstm_lds, stream>>>(gates_f, gates_b, Whh_f, Whh_b,
                                                       hf, hb, hbf_f, hbf_b, counters);

    // concat -> bf16, then output projection GEMM
    concat_bf16<<<(unsigned)LB, 256, 0, stream>>>(hf, hb, hcat);
    gemm_bf16<Hx><<<dim3(KK / 128, (unsigned)(LB / 16)), 256, 0, stream>>>(
        hcat, woutb, bout, nullptr, scores, (int)LB, KK);

    // gold score partials, CRF forward partials, final combine
    gold_kernel<<<128, 256, 0, stream>>>(scores, targ, masks, gold_part);
    crf_kernel<<<Bx, 1024, 0, stream>>>(scores, masks, crf_part);
    finalize_kernel<<<1, 128, 0, stream>>>(gold_part, crf_part, out);
}
